// GCNAggregator_50354196578649
// MI455X (gfx1250) — matrix operand order repaired and checked by reading, not compiled, not run on hardware
//
#include <hip/hip_runtime.h>
#include <math.h>

#define Bd   512
#define Nd   128
#define Ed   1024
#define Cd   1000
#define EQd  256
#define Md   (Bd*Nd)
#define EPSd 1e-5f

typedef __bf16 bf16;
typedef __attribute__((ext_vector_type(16))) __bf16 v16bf;
typedef __attribute__((ext_vector_type(4)))  __bf16 v4bf;
typedef __attribute__((ext_vector_type(2)))  __bf16 v2bf;
typedef __attribute__((ext_vector_type(8)))  float  v8f;

union FragBF {
    uint4 q[2];
    v16bf v;
};

// A-matrix frag (16x32 bf16): lane<16 -> K chunks [kk+0..7] & [kk+16..23];
// lane>=16 -> [kk+8..15] & [kk+24..31]  (ISA 7.12.2, 16-bit A 16x32)
__device__ __forceinline__ v16bf loadA_bf16(const bf16* rowbase, int kk, int h) {
    FragBF f;
    f.q[0] = *(const uint4*)(rowbase + kk + h * 8);
    f.q[1] = *(const uint4*)(rowbase + kk + 16 + h * 8);
    return f.v;
}

// B-matrix frag (32x16 bf16), source is B^T row-major [n][k]:
// lane<16 -> K [kk+0..7] & [kk+16..23]; lane>=16 -> K [kk+8..15] & [kk+24..31]
__device__ __forceinline__ v16bf loadB_bf16(const bf16* rowbase, int kk, int h) {
    FragBF f;
    f.q[0] = *(const uint4*)(rowbase + kk + h * 8);
    f.q[1] = *(const uint4*)(rowbase + kk + 16 + h * 8);
    return f.v;
}

#define WMMA_BF16(A, B, C) \
    __builtin_amdgcn_wmma_f32_16x16x32_bf16(false, (A), false, (B), (short)0, (C), false, false)

// CDNA5 async global->LDS copy (tracked by ASYNCcnt).  lds_off = low 32 bits of
// the generic pointer (LDS aperture is addr[63:32], offset is addr[31:0], ISA 10.2)
__device__ __forceinline__ void async_b128(unsigned lds_off, const bf16* g) {
    asm volatile("global_load_async_to_lds_b128 %0, %1, off"
                 :: "v"(lds_off), "v"(g) : "memory");
}
__device__ __forceinline__ void wait_async() {
    asm volatile("s_wait_asynccnt 0x0" ::: "memory");
}
__device__ __forceinline__ unsigned lds_off(const void* p) {
    return (unsigned)(uintptr_t)p;
}

// ---------------- K0: fold Wq/Wk means, bias means, Wp sum ----------------
__global__ void k0_prep(const float* __restrict__ Wq, const float* __restrict__ bq,
                        const float* __restrict__ Wk, const float* __restrict__ bk,
                        const float* __restrict__ Wp,
                        float* __restrict__ wqb, float* __restrict__ wkb,
                        float* __restrict__ scal) {
    __shared__ float red[256];
    int t = threadIdx.x;
    for (int c = t; c < Ed; c += 256) {
        float sq = 0.f, sk = 0.f;
        for (int q = 0; q < EQd; ++q) { sq += Wq[q * Ed + c]; sk += Wk[q * Ed + c]; }
        wqb[c] = sq * (1.0f / EQd);
        wkb[c] = sk * (1.0f / EQd);
    }
    red[t] = bq[t]; __syncthreads();
    for (int s = 128; s > 0; s >>= 1) { if (t < s) red[t] += red[t + s]; __syncthreads(); }
    if (t == 0) scal[0] = red[0] * (1.0f / EQd);
    __syncthreads();
    red[t] = bk[t]; __syncthreads();
    for (int s = 128; s > 0; s >>= 1) { if (t < s) red[t] += red[t + s]; __syncthreads(); }
    if (t == 0) scal[1] = red[0] * (1.0f / EQd);
    __syncthreads();
    red[t] = (t < Nd) ? Wp[t] : 0.f; __syncthreads();
    for (int s = 128; s > 0; s >>= 1) { if (t < s) red[t] += red[t + s]; __syncthreads(); }
    if (t == 0) scal[2] = red[0];
}

// ---------------- K0b: convert W1, Wc (padded to 1024 rows) to bf16 ----------------
__global__ void k0_convert(const float* __restrict__ W1, const float* __restrict__ Wc,
                           bf16* __restrict__ W1b, bf16* __restrict__ Wcb) {
    int idx = blockIdx.x * 256 + threadIdx.x;   // 2*Ed*Ed total
    if (idx < Ed * Ed) {
        W1b[idx] = (bf16)W1[idx];
    } else {
        int j = idx - Ed * Ed;
        int c = j >> 10, e = j & 1023;
        Wcb[j] = (c < Cd) ? (bf16)Wc[c * Ed + e] : (bf16)0.0f;
    }
}

// ---------------- K1: q1/k1 dots with folded weights + x -> bf16 copy ----------------
__global__ void __launch_bounds__(256) k1_qk(const float* __restrict__ x,
                                             const float* __restrict__ wqb,
                                             const float* __restrict__ wkb,
                                             const float* __restrict__ scal,
                                             float* __restrict__ q1, float* __restrict__ k1,
                                             bf16* __restrict__ xb) {
    int t = threadIdx.x, w = t >> 5, lane = t & 31;
    int row = blockIdx.x * 8 + w;
    const float4* xp  = (const float4*)(x + (size_t)row * Ed);
    const float4* wq4 = (const float4*)wqb;
    const float4* wk4 = (const float4*)wkb;
    bf16* xo = xb + (size_t)row * Ed;
    float sq = 0.f, sk = 0.f;
#pragma unroll
    for (int i = 0; i < 8; ++i) {
        int c = lane + i * 32;
        float4 xv = xp[c], qv = wq4[c], kv = wk4[c];
        sq += xv.x * qv.x + xv.y * qv.y + xv.z * qv.z + xv.w * qv.w;
        sk += xv.x * kv.x + xv.y * kv.y + xv.z * kv.z + xv.w * kv.w;
        v4bf o;
        o[0] = (bf16)xv.x; o[1] = (bf16)xv.y; o[2] = (bf16)xv.z; o[3] = (bf16)xv.w;
        *(v4bf*)(xo + c * 4) = o;
    }
#pragma unroll
    for (int m = 16; m > 0; m >>= 1) {
        sq += __shfl_xor(sq, m, 32);
        sk += __shfl_xor(sk, m, 32);
    }
    if (lane == 0) { q1[row] = sq + scal[0]; k1[row] = sk + scal[1]; }
}

// ---------------- K2: A1T[b,m,n] = adj1[n,m] + tanh(q1[b,n]-k1[b,m])*alpha (bf16) ----------------
__global__ void k2_attn(const float* __restrict__ q1, const float* __restrict__ k1,
                        const float* __restrict__ adj1, const float* __restrict__ alpha,
                        bf16* __restrict__ A1T) {
    int idx = blockIdx.x * 256 + threadIdx.x;   // B*N*N/2 threads, 2 n per thread
    int b = idx >> 13;
    int r = idx & 8191;
    int m = r >> 6;
    int n0 = (r & 63) << 1;
    float a  = alpha[0];
    float kv = k1[b * Nd + m];
    float va = adj1[n0 * Nd + m]       + tanhf(q1[b * Nd + n0]     - kv) * a;
    float vb = adj1[(n0 + 1) * Nd + m] + tanhf(q1[b * Nd + n0 + 1] - kv) * a;
    v2bf o; o[0] = (bf16)va; o[1] = (bf16)vb;
    *(v2bf*)(A1T + (size_t)b * (Nd * Nd) + m * Nd + n0) = o;
}

// ---------------- K3: Y = xb @ W1^T + b1 ----------------
// LDS-tiled, double-buffered via GLOBAL_LOAD_ASYNC_TO_LDS_B128 (ASYNCcnt).
// Workgroup tile 64m x 256e, K-slab 32; wave tile 32x64 (8 wmma / k-step).
// grid: (Ed/256, Md/64) -- e-tiles fastest so blocks sharing rows hit L2
#define KSLAB 32
#define LSTR  40              // padded LDS row stride (elems), 16B-aligned, conflict-free
__global__ void __launch_bounds__(256) k3_gemm1(const bf16* __restrict__ xb,
                                                const bf16* __restrict__ W1b,
                                                const float* __restrict__ b1,
                                                bf16* __restrict__ Yb) {
    __shared__ bf16 Ab[2][64 * LSTR];     // A slab: [m][k]
    __shared__ bf16 Bb[2][256 * LSTR];    // B slab: [n][k] (row-major along K)
    int t = threadIdx.x, w = t >> 5, lane = t & 31;
    int h = lane >> 4, r16 = lane & 15;
    int e_base = blockIdx.x * 256;
    int m_base = blockIdx.y * 64;
    // staging assignment: 1 A chunk + 4 B chunks (16B each) per thread per slab
    int arow = t >> 2, ako = (t & 3) * 8;
    const bf16* aG = xb + (size_t)(m_base + arow) * Ed + ako;
    int wm = w & 1, we = w >> 1;
    int m0 = wm * 32, e0 = we * 64;
    v8f acc[2][4] = {};

    auto stage = [&](int kb, int buf) {
        async_b128(lds_off(&Ab[buf][arow * LSTR + ako]), aG + kb);
#pragma unroll
        for (int j = 0; j < 4; ++j) {
            int c = t + 256 * j;
            int brow = c >> 2, bko = (c & 3) * 8;
            async_b128(lds_off(&Bb[buf][brow * LSTR + bko]),
                       W1b + (size_t)(e_base + brow) * Ed + kb + bko);
        }
    };

    stage(0, 0);
    wait_async();
    __syncthreads();
#pragma unroll 2
    for (int s = 0; s < Ed / KSLAB; ++s) {
        int buf = s & 1;
        if (s + 1 < Ed / KSLAB) stage((s + 1) * KSLAB, buf ^ 1);
        const bf16* Ap = &Ab[buf][0];
        const bf16* Bp = &Bb[buf][0];
        v16bf a0 = loadA_bf16(Ap + (m0 + r16) * LSTR, 0, h);
        v16bf a1 = loadA_bf16(Ap + (m0 + 16 + r16) * LSTR, 0, h);
#pragma unroll
        for (int fi = 0; fi < 4; ++fi) {
            v16bf bf = loadB_bf16(Bp + (e0 + fi * 16 + r16) * LSTR, 0, h);
            acc[0][fi] = WMMA_BF16(a0, bf, acc[0][fi]);
            acc[1][fi] = WMMA_BF16(a1, bf, acc[1][fi]);
        }
        if (s + 1 < Ed / KSLAB) wait_async();
        __syncthreads();
    }
#pragma unroll
    for (int mi = 0; mi < 2; ++mi) {
#pragma unroll
        for (int fi = 0; fi < 4; ++fi) {
            int n = e_base + e0 + fi * 16 + r16;
            float bias = b1[n];
            int mrow = m_base + m0 + mi * 16 + h * 8;
#pragma unroll
            for (int i = 0; i < 8; ++i)
                Yb[(size_t)(mrow + i) * Ed + n] = (bf16)(acc[mi][fi][i] + bias);
        }
    }
}

// ---------------- K4: ZT_b = A1T_b @ Y_b, reduced in-place (sum, sum^2, ·Wp) ----------------
__global__ void __launch_bounds__(256) k4_gemm2(const bf16* __restrict__ Yb,
                                                const bf16* __restrict__ A1T,
                                                const float* __restrict__ Wp,
                                                float* __restrict__ SsumB,
                                                float* __restrict__ SsqB,
                                                float* __restrict__ P) {
    __shared__ unsigned short YT[128 * 136];    // Y tile transposed: [e][n], padded stride
    __shared__ float partS[4][128];
    __shared__ float partQ[4][128];
    __shared__ float partP[4][128];
    __shared__ float wp_s[128];
    int b = blockIdx.x;
    int e_base = blockIdx.y * 128;
    int t = threadIdx.x;
    {   // stage + transpose Y[b, n, e_base..+128) into LDS
        int n = t >> 1, eo = (t & 1) * 64;
        const unsigned short* yrow =
            (const unsigned short*)(Yb + (size_t)(b * Nd + n) * Ed + e_base + eo);
#pragma unroll
        for (int j = 0; j < 16; ++j) {
            uint2 d = *(const uint2*)(yrow + j * 4);
            int e4 = eo + j * 4;
            YT[(e4 + 0) * 136 + n] = (unsigned short)(d.x & 0xFFFFu);
            YT[(e4 + 1) * 136 + n] = (unsigned short)(d.x >> 16);
            YT[(e4 + 2) * 136 + n] = (unsigned short)(d.y & 0xFFFFu);
            YT[(e4 + 3) * 136 + n] = (unsigned short)(d.y >> 16);
        }
        if (t < Nd) wp_s[t] = Wp[t];
    }
    __syncthreads();

    int w = t >> 5, lane = t & 31, h = lane >> 4, r16 = lane & 15;
    int wm = w & 3, we = w >> 2;
    int m0 = wm * 32;
    const bf16* a0p = A1T + (size_t)b * (Nd * Nd) + (m0 + r16) * Nd;
    const bf16* a1p = a0p + 16 * Nd;
    v8f acc[2][4] = {};
#pragma unroll
    for (int kk = 0; kk < Nd; kk += 32) {
        v16bf a0 = loadA_bf16(a0p, kk, h);
        v16bf a1 = loadA_bf16(a1p, kk, h);
#pragma unroll
        for (int fi = 0; fi < 4; ++fi) {
            const bf16* bpl = (const bf16*)&YT[(we * 64 + fi * 16 + r16) * 136];
            v16bf bf = loadB_bf16(bpl, kk, h);
            acc[0][fi] = WMMA_BF16(a0, bf, acc[0][fi]);
            acc[1][fi] = WMMA_BF16(a1, bf, acc[1][fi]);
        }
    }
    // deterministic reduction: each wave owns disjoint [wm][el] slots
#pragma unroll
    for (int fi = 0; fi < 4; ++fi) {
        float s1 = 0.f, s2 = 0.f, sp = 0.f;
#pragma unroll
        for (int mi = 0; mi < 2; ++mi) {
            int mb = m0 + mi * 16 + h * 8;
#pragma unroll
            for (int i = 0; i < 8; ++i) {
                float v = acc[mi][fi][i];
                s1 += v; s2 += v * v; sp += v * wp_s[mb + i];
            }
        }
        s1 += __shfl_xor(s1, 16, 32);
        s2 += __shfl_xor(s2, 16, 32);
        sp += __shfl_xor(sp, 16, 32);
        if (h == 0) {
            int el = we * 64 + fi * 16 + r16;
            partS[wm][el] = s1; partQ[wm][el] = s2; partP[wm][el] = sp;
        }
    }
    __syncthreads();
    if (t < 128) {
        float s1 = partS[0][t] + partS[1][t] + partS[2][t] + partS[3][t];
        float s2 = partQ[0][t] + partQ[1][t] + partQ[2][t] + partQ[3][t];
        float sp = partP[0][t] + partP[1][t] + partP[2][t] + partP[3][t];
        int eg = e_base + t;
        SsumB[(size_t)b * Ed + eg] = s1;
        SsqB [(size_t)b * Ed + eg] = s2;
        P    [(size_t)b * Ed + eg] = sp;
    }
}

// ---------------- K4b: deterministic BN-stat reduce over batches -> affine per channel ----------------
__global__ void k4b_bnstats(const float* __restrict__ SsumB, const float* __restrict__ SsqB,
                            const float* __restrict__ gamma, const float* __restrict__ beta,
                            const float* __restrict__ scal, const float* __restrict__ bp,
                            float* __restrict__ scaleA, float* __restrict__ shiftA) {
    int e = blockIdx.x * 256 + threadIdx.x;
    float s1 = 0.f, s2 = 0.f;
    for (int b = 0; b < Bd; ++b) {
        s1 += SsumB[(size_t)b * Ed + e];
        s2 += SsqB [(size_t)b * Ed + e];
    }
    const float inv = 1.0f / (float)(Bd * Nd);
    float mu  = s1 * inv;
    float var = s2 * inv - mu * mu;
    float sc  = gamma[e] * rsqrtf(var + EPSd);
    float wps = scal[2];
    scaleA[e] = sc;
    shiftA[e] = beta[e] * wps + bp[0] - sc * mu * wps;
}

// ---------------- K5: H[b,e] = scale[e]*P[b,e] + shift[e] -> bf16 ----------------
__global__ void k5_pool(const float* __restrict__ P, const float* __restrict__ scaleA,
                        const float* __restrict__ shiftA, bf16* __restrict__ Hb) {
    int idx = blockIdx.x * 256 + threadIdx.x;
    int e = idx & (Ed - 1);
    Hb[idx] = (bf16)(scaleA[e] * P[idx] + shiftA[e]);
}

// ---------------- K6: out = H @ Wc^T + bc   (bf16 WMMA, N padded to 1024) ----------------
__global__ void __launch_bounds__(256) k6_cls(const bf16* __restrict__ Hb,
                                              const bf16* __restrict__ Wcb,
                                              const float* __restrict__ bc,
                                              float* __restrict__ out) {
    int t = threadIdx.x, w = t >> 5, lane = t & 31;
    int h = lane >> 4, r16 = lane & 15;
    int m0 = blockIdx.y * 32 + (w & 1) * 16;
    int e0 = blockIdx.x * 256 + (w >> 1) * 64;
    const bf16* ap  = Hb + (size_t)(m0 + r16) * Ed;
    const bf16* bp0 = Wcb + (size_t)(e0 + r16) * Ed;
    v8f acc[4] = {};
    v16bf aC  = loadA_bf16(ap, 0, h);
    v16bf bC0 = loadB_bf16(bp0,           0, h);
    v16bf bC1 = loadB_bf16(bp0 + 16 * Ed, 0, h);
    v16bf bC2 = loadB_bf16(bp0 + 32 * Ed, 0, h);
    v16bf bC3 = loadB_bf16(bp0 + 48 * Ed, 0, h);
#pragma unroll
    for (int kk = 0; kk < Ed; kk += 32) {
        v16bf aN, bN0, bN1, bN2, bN3;
        const bool more = (kk + 32) < Ed;
        if (more) {
            aN  = loadA_bf16(ap, kk + 32, h);
            bN0 = loadB_bf16(bp0,           kk + 32, h);
            bN1 = loadB_bf16(bp0 + 16 * Ed, kk + 32, h);
            bN2 = loadB_bf16(bp0 + 32 * Ed, kk + 32, h);
            bN3 = loadB_bf16(bp0 + 48 * Ed, kk + 32, h);
        }
        acc[0] = WMMA_BF16(aC, bC0, acc[0]);
        acc[1] = WMMA_BF16(aC, bC1, acc[1]);
        acc[2] = WMMA_BF16(aC, bC2, acc[2]);
        acc[3] = WMMA_BF16(aC, bC3, acc[3]);
        if (more) { aC = aN; bC0 = bN0; bC1 = bN1; bC2 = bN2; bC3 = bN3; }
    }
#pragma unroll
    for (int fi = 0; fi < 4; ++fi) {
        int c = e0 + fi * 16 + r16;
        if (c < Cd) {
            float bias = bc[c];
            int mrow = m0 + h * 8;
#pragma unroll
            for (int i = 0; i < 8; ++i)
                out[(size_t)(mrow + i) * Cd + c] = acc[fi][i] + bias;
        }
    }
}

extern "C" void kernel_launch(void* const* d_in, const int* in_sizes, int n_in,
                              void* d_out, int out_size, void* d_ws, size_t ws_size,
                              hipStream_t stream) {
    (void)in_sizes; (void)n_in; (void)out_size; (void)ws_size;
    const float* x     = (const float*)d_in[0];
    const float* adj1  = (const float*)d_in[1];
    const float* alpha = (const float*)d_in[2];
    const float* Wq    = (const float*)d_in[3];
    const float* bq    = (const float*)d_in[4];
    const float* Wk    = (const float*)d_in[5];
    const float* bk    = (const float*)d_in[6];
    const float* W1    = (const float*)d_in[7];
    const float* b1    = (const float*)d_in[8];
    const float* gamma = (const float*)d_in[9];
    const float* beta  = (const float*)d_in[10];
    const float* Wp    = (const float*)d_in[11];
    const float* bp    = (const float*)d_in[12];
    const float* Wc    = (const float*)d_in[13];
    const float* bc    = (const float*)d_in[14];
    float* out = (float*)d_out;

    char* ws = (char*)d_ws;
    size_t off = 0;
    auto alloc = [&](size_t bytes) -> void* {
        void* p = ws + off;
        off += (bytes + 255) & ~(size_t)255;
        return p;
    };
    bf16* xb    = (bf16*)alloc((size_t)Md * Ed * 2 + 256);   // 128 MiB (+pad)
    bf16* Yb    = (bf16*)alloc((size_t)Md * Ed * 2 + 256);   // 128 MiB (+pad)
    bf16* W1b   = (bf16*)alloc((size_t)Ed * Ed * 2 + 256);
    bf16* Wcb   = (bf16*)alloc((size_t)Ed * Ed * 2 + 256);
    bf16* A1T   = (bf16*)alloc((size_t)Bd * Nd * Nd * 2);
    float* q1     = (float*)alloc((size_t)Md * 4);
    float* k1v    = (float*)alloc((size_t)Md * 4);
    float* wqb    = (float*)alloc(Ed * 4);
    float* wkb    = (float*)alloc(Ed * 4);
    float* scal   = (float*)alloc(256);
    float* SsumB  = (float*)alloc((size_t)Bd * Ed * 4);
    float* SsqB   = (float*)alloc((size_t)Bd * Ed * 4);
    float* Pbuf   = (float*)alloc((size_t)Bd * Ed * 4);
    float* scaleA = (float*)alloc(Ed * 4);
    float* shiftA = (float*)alloc(Ed * 4);
    bf16* Hb    = (bf16*)alloc((size_t)Bd * Ed * 2 + 256);

    k0_prep<<<1, 256, 0, stream>>>(Wq, bq, Wk, bk, Wp, wqb, wkb, scal);
    k0_convert<<<(2 * Ed * Ed) / 256, 256, 0, stream>>>(W1, Wc, W1b, Wcb);
    k1_qk<<<Md / 8, 256, 0, stream>>>(x, wqb, wkb, scal, q1, k1v, xb);
    k2_attn<<<(Bd * Nd * Nd / 2) / 256, 256, 0, stream>>>(q1, k1v, adj1, alpha, A1T);
    k3_gemm1<<<dim3(Ed / 256, Md / 64), 256, 0, stream>>>(xb, W1b, b1, Yb);
    k4_gemm2<<<dim3(Bd, Ed / 128), 256, 0, stream>>>(Yb, A1T, Wp, SsumB, SsqB, Pbuf);
    k4b_bnstats<<<Ed / 256, 256, 0, stream>>>(SsumB, SsqB, gamma, beta, scal, bp, scaleA, shiftA);
    k5_pool<<<(Bd * Ed) / 256, 256, 0, stream>>>(Pbuf, scaleA, shiftA, Hb);
    k6_cls<<<dim3(Ed / 256, Bd / 32), 256, 0, stream>>>(Hb, Wcb, bc, out);
}
